// HazardGNN_43636867727631
// MI455X (gfx1250) — matrix operand order repaired and checked by reading, not compiled, not run on hardware
//
#include <hip/hip_runtime.h>

// ---------------------------------------------------------------------------
// Types for CDNA5 WMMA
// ---------------------------------------------------------------------------
typedef __bf16 v16bf __attribute__((ext_vector_type(16)));
typedef float  v8f   __attribute__((ext_vector_type(8)));

#define D_NODE 64
#define D_MSG  64
#define D_IN   132
#define K_PAD  160   // 132 padded to 5 K-tiles of 32
#define NLAYER 3

// ---------------------------------------------------------------------------
// Fragment loaders (contiguous K order, same for A and B, wave32)
// ---------------------------------------------------------------------------

// A fragment (16-bit, 16x32 MxK) from an LDS tile of leading dim `ld`.
// lane L<16 holds row M=L, K = kt*32 + 0..15;
// lane L>=16 holds row M=L-16, K = kt*32 + 16..31.
__device__ __forceinline__ v16bf load_afrag(const __bf16* t, int ld, int kt, int lane) {
  int row  = lane & 15;
  int half = (lane >> 4) & 1;
  const __bf16* p = t + row * ld + kt * 32 + half * 16;
  v16bf a;
#pragma unroll
  for (int i = 0; i < 8; ++i) {
    int off = 2 * i;  // contiguous K base+0..15
    a[2 * i]     = p[off];
    a[2 * i + 1] = p[off + 1];
  }
  return a;
}

// B fragment from pre-fragmented weights: layout [nt][kt][lane][16] bf16,
// 32 bytes contiguous per lane -> wide loads.
__device__ __forceinline__ v16bf load_bfrag(const __bf16* f, int nt, int kt, int KT, int lane) {
  const __bf16* p = f + ((((nt * KT) + kt) * 32 + lane) << 4);
  return *(const v16bf*)p;
}

__device__ __forceinline__ float fast_sigmoid(float x) {
  return 1.0f / (1.0f + __expf(-x));
}

// Pack 4 f32 -> 4 bf16 in a uint2 (for 8-byte LDS stores)
__device__ __forceinline__ uint2 pack4_bf16(float x, float y, float z, float w) {
  union { __bf16 b[4]; uint2 u; } t;
  t.b[0] = (__bf16)x; t.b[1] = (__bf16)y; t.b[2] = (__bf16)z; t.b[3] = (__bf16)w;
  return t.u;
}

// ---------------------------------------------------------------------------
// Weight pre-fragmentation: f32 [out][in] -> bf16 WMMA-B per-lane layout.
//   b[j] (lane) = W[n][kbase + j],  kbase = kt*32 + ((lane>=16)?16:0),
//   n = nt*16 + (lane&15).  K padded with zeros beyond valid K.
// w1f: [3][4nt][5kt][32][16], w2f: [3][4nt][2kt][32][16], hwf: [4nt][5kt][32][16]
// ---------------------------------------------------------------------------
__global__ void prep_weights_kernel(const float* __restrict__ msg_w1,
                                    const float* __restrict__ msg_w2,
                                    const float* __restrict__ head_w1,
                                    __bf16* __restrict__ w1f,
                                    __bf16* __restrict__ w2f,
                                    __bf16* __restrict__ hwf) {
  int idx = blockIdx.x * blockDim.x + threadIdx.x;
  const int W1N = NLAYER * 4 * 5 * 32 * 16;  // 30720
  const int W2N = NLAYER * 4 * 2 * 32 * 16;  // 12288
  const int HWN = 4 * 5 * 32 * 16;           // 10240
  if (idx < W1N) {
    int t = idx;
    int l  = t / (4 * 5 * 32 * 16); t %= (4 * 5 * 32 * 16);
    int nt = t / (5 * 32 * 16);     t %= (5 * 32 * 16);
    int kt = t / (32 * 16);         t %= (32 * 16);
    int lane = t >> 4, j = t & 15;
    int k = kt * 32 + ((lane >> 4) & 1) * 16 + j;
    int n = nt * 16 + (lane & 15);
    float v = (k < D_IN) ? msg_w1[(size_t)l * D_MSG * D_IN + n * D_IN + k] : 0.0f;
    w1f[idx] = (__bf16)v;
  } else if (idx < W1N + W2N) {
    int t = idx - W1N;
    int l  = t / (4 * 2 * 32 * 16); int r = t % (4 * 2 * 32 * 16);
    int nt = r / (2 * 32 * 16);     r %= (2 * 32 * 16);
    int kt = r / (32 * 16);         r %= (32 * 16);
    int lane = r >> 4, j = r & 15;
    int k = kt * 32 + ((lane >> 4) & 1) * 16 + j;
    int n = nt * 16 + (lane & 15);
    w2f[t] = (__bf16)msg_w2[(size_t)l * 64 * 64 + n * 64 + k];
  } else if (idx < W1N + W2N + HWN) {
    int t = idx - W1N - W2N;
    int nt = t / (5 * 32 * 16); int r = t % (5 * 32 * 16);
    int kt = r / (32 * 16);     r %= (32 * 16);
    int lane = r >> 4, j = r & 15;
    int k = kt * 32 + ((lane >> 4) & 1) * 16 + j;
    int n = nt * 16 + (lane & 15);
    float v = (k < D_IN) ? head_w1[n * D_IN + k] : 0.0f;
    hwf[t] = (__bf16)v;
  }
}

// ---------------------------------------------------------------------------
// h init and agg zero
// ---------------------------------------------------------------------------
__global__ void init_h_kernel(float* __restrict__ h, const float* __restrict__ emb,
                              const int* __restrict__ kinds, int total) {
  int idx = blockIdx.x * blockDim.x + threadIdx.x;
  if (idx < total) {
    int node = idx >> 6, d = idx & 63;
    h[idx] = emb[(size_t)kinds[node] * 64 + d];
  }
}

__global__ void zero_kernel(float* __restrict__ p, int total) {
  int idx = blockIdx.x * blockDim.x + threadIdx.x;
  if (idx < total) p[idx] = 0.0f;
}

// ---------------------------------------------------------------------------
// Vectorized x-tile staging (shared by msg & head kernels):
// lanes 0-15 load the src row (cols 0..63) of edge e0+lane as 16 float4s,
// lanes 16-31 load the dst row (cols 64..127); half 0 also stages edge_attr
// (cols 128..131), half 1 writes the zero pad (cols 132..159).
// All LDS stores are 8-byte packed 4xbf16.
// ---------------------------------------------------------------------------
__device__ __forceinline__ void stage_x_tile(__bf16* xs, const float* __restrict__ h,
                                             const int* __restrict__ eidx,
                                             const float* __restrict__ eattr,
                                             long e0, int E, int lane) {
  int half = lane >> 4;     // 0 = src row, 1 = dst row
  int r    = lane & 15;
  long e   = e0 + r;
  bool valid = (e < E);
  int node = valid ? eidx[(long)half * E + e] : 0;
  const float4* hp = (const float4*)(h + (size_t)node * 64);
  uint2* row = (uint2*)(xs + r * K_PAD + half * 64);
#pragma unroll
  for (int q = 0; q < 16; ++q) {
    float4 v = valid ? hp[q] : make_float4(0.f, 0.f, 0.f, 0.f);
    row[q] = pack4_bf16(v.x, v.y, v.z, v.w);
  }
  if (half == 0) {
    float4 v = valid ? *(const float4*)(eattr + e * 4) : make_float4(0.f, 0.f, 0.f, 0.f);
    *(uint2*)(xs + r * K_PAD + 128) = pack4_bf16(v.x, v.y, v.z, v.w);
  } else {
    uint2 z; z.x = 0u; z.y = 0u;
    uint2* zp = (uint2*)(xs + r * K_PAD + 132);
#pragma unroll
    for (int q = 0; q < 7; ++q) zp[q] = z;
  }
}

// ---------------------------------------------------------------------------
// Fused edge-message kernel (one wave = 16 edges):
//   x = [h[src] | h[dst] | eattr]  (bf16 tile in LDS)
//   hid = silu(x @ W1^T + b1)   -- 5x4 WMMA bf16
//   m   = hid @ W2^T + b2       -- 2x4 WMMA bf16
//   atomicAdd agg[dst] += m     (L2-resident)
// ---------------------------------------------------------------------------
__global__ __launch_bounds__(256) void edge_msg_kernel(
    const float* __restrict__ h, const int* __restrict__ eidx,
    const float* __restrict__ eattr, float* __restrict__ agg,
    const __bf16* __restrict__ w1f, const float* __restrict__ b1,
    const __bf16* __restrict__ w2f, const float* __restrict__ b2, int E) {
  __shared__ __bf16 lds[8 * 16 * K_PAD];   // 40 KB: 8 waves x [16][160] bf16
  int wave = threadIdx.x >> 5;
  int lane = threadIdx.x & 31;
  __bf16* xs = lds + wave * 16 * K_PAD;
  long tile = (long)blockIdx.x * 8 + wave;
  long e0 = tile * 16;

  stage_x_tile(xs, h, eidx, eattr, e0, E, lane);
  __syncthreads();

  // ---- A fragments for GEMM1 (whole [16,160] tile in regs) ----
  v16bf a[5];
#pragma unroll
  for (int kt = 0; kt < 5; ++kt) a[kt] = load_afrag(xs, K_PAD, kt, lane);
  __syncthreads();  // xs region will be reused for hidden tile

  int col   = lane & 15;
  int rbase = ((lane >> 4) & 1) * 8;

  // destination node ids for this lane's 8 C rows (shared across nt tiles)
  int dsts[8];
#pragma unroll
  for (int i = 0; i < 8; ++i) {
    long e = e0 + rbase + i;
    dsts[i] = (e < E) ? eidx[(long)E + e] : -1;
  }

  // ---- GEMM1 + SiLU, restripe hidden [16,64] into LDS ----
#pragma unroll
  for (int nt = 0; nt < 4; ++nt) {
    float bv = b1[nt * 16 + col];
    v8f c = {bv, bv, bv, bv, bv, bv, bv, bv};
#pragma unroll
    for (int kt = 0; kt < 5; ++kt) {
      v16bf b = load_bfrag(w1f, nt, kt, 5, lane);
      c = __builtin_amdgcn_wmma_f32_16x16x32_bf16(false, a[kt], false, b,
                                                  (short)0, c, false, false);
    }
#pragma unroll
    for (int i = 0; i < 8; ++i) {
      float v = c[i];
      v = v * fast_sigmoid(v);  // SiLU
      xs[(rbase + i) * 64 + nt * 16 + col] = (__bf16)v;
    }
  }
  __syncthreads();

  // ---- GEMM2 + scatter ----
  v16bf a2[2];
#pragma unroll
  for (int kt = 0; kt < 2; ++kt) a2[kt] = load_afrag(xs, 64, kt, lane);
#pragma unroll
  for (int nt = 0; nt < 4; ++nt) {
    float bv = b2[nt * 16 + col];
    v8f c = {bv, bv, bv, bv, bv, bv, bv, bv};
#pragma unroll
    for (int kt = 0; kt < 2; ++kt) {
      v16bf b = load_bfrag(w2f, nt, kt, 2, lane);
      c = __builtin_amdgcn_wmma_f32_16x16x32_bf16(false, a2[kt], false, b,
                                                  (short)0, c, false, false);
    }
#pragma unroll
    for (int i = 0; i < 8; ++i) {
      if (dsts[i] >= 0) {
        __hip_atomic_fetch_add(&agg[(size_t)dsts[i] * 64 + nt * 16 + col], c[i],
                               __ATOMIC_RELAXED, __HIP_MEMORY_SCOPE_AGENT);
      }
    }
  }
}

// ---------------------------------------------------------------------------
// GRU cell (torch layout), f32 VALU. 64 threads per node, 4 nodes/block.
// ---------------------------------------------------------------------------
__global__ __launch_bounds__(256) void gru_kernel(
    float* __restrict__ h, const float* __restrict__ agg,
    const float* __restrict__ wih, const float* __restrict__ whh,
    const float* __restrict__ bih, const float* __restrict__ bhh, int N) {
  __shared__ float sa[4][64];
  __shared__ float sh[4][64];
  int g = threadIdx.x >> 6;
  int j = threadIdx.x & 63;
  int node = blockIdx.x * 4 + g;
  if (node < N) {
    sa[g][j] = agg[(size_t)node * 64 + j];
    sh[g][j] = h[(size_t)node * 64 + j];
  }
  __syncthreads();
  if (node >= N) return;
  float gi[3], gh[3];
#pragma unroll
  for (int t = 0; t < 3; ++t) {
    int row = t * 64 + j;
    float ai = bih[row], ah = bhh[row];
    const float* wi = wih + (size_t)row * 64;
    const float* wh = whh + (size_t)row * 64;
#pragma unroll 8
    for (int k = 0; k < 64; ++k) {
      ai = fmaf(wi[k], sa[g][k], ai);
      ah = fmaf(wh[k], sh[g][k], ah);
    }
    gi[t] = ai; gh[t] = ah;
  }
  float r  = fast_sigmoid(gi[0] + gh[0]);
  float z  = fast_sigmoid(gi[1] + gh[1]);
  float nn = tanhf(gi[2] + r * gh[2]);
  h[(size_t)node * 64 + j] = (1.0f - z) * nn + z * sh[g][j];
}

// ---------------------------------------------------------------------------
// Edge head: hid = silu(x @ Wh1^T + b1) via WMMA, logit = hid . w2 + b2,
// out = sigmoid(logit)
// ---------------------------------------------------------------------------
__global__ __launch_bounds__(256) void edge_head_kernel(
    const float* __restrict__ h, const int* __restrict__ eidx,
    const float* __restrict__ eattr, float* __restrict__ out,
    const __bf16* __restrict__ hwf, const float* __restrict__ b1,
    const float* __restrict__ w2, const float* __restrict__ b2, int E) {
  __shared__ __bf16 lds[8 * 16 * K_PAD];
  int wave = threadIdx.x >> 5;
  int lane = threadIdx.x & 31;
  __bf16* xs = lds + wave * 16 * K_PAD;
  long tile = (long)blockIdx.x * 8 + wave;
  long e0 = tile * 16;

  stage_x_tile(xs, h, eidx, eattr, e0, E, lane);
  __syncthreads();

  v16bf a[5];
#pragma unroll
  for (int kt = 0; kt < 5; ++kt) a[kt] = load_afrag(xs, K_PAD, kt, lane);
  __syncthreads();

  int col   = lane & 15;
  int rbase = ((lane >> 4) & 1) * 8;
#pragma unroll
  for (int nt = 0; nt < 4; ++nt) {
    float bv = b1[nt * 16 + col];
    v8f c = {bv, bv, bv, bv, bv, bv, bv, bv};
#pragma unroll
    for (int kt = 0; kt < 5; ++kt) {
      v16bf b = load_bfrag(hwf, nt, kt, 5, lane);
      c = __builtin_amdgcn_wmma_f32_16x16x32_bf16(false, a[kt], false, b,
                                                  (short)0, c, false, false);
    }
#pragma unroll
    for (int i = 0; i < 8; ++i) {
      float v = c[i];
      v = v * fast_sigmoid(v);
      xs[(rbase + i) * 64 + nt * 16 + col] = (__bf16)v;
    }
  }
  __syncthreads();

  if (lane < 16) {
    long e = e0 + lane;
    if (e < E) {
      float acc = b2[0];
#pragma unroll 8
      for (int n2 = 0; n2 < 64; ++n2) acc += (float)xs[lane * 64 + n2] * w2[n2];
      out[e] = fast_sigmoid(acc);
    }
  }
}

// ---------------------------------------------------------------------------
// Launch
// ---------------------------------------------------------------------------
extern "C" void kernel_launch(void* const* d_in, const int* in_sizes, int n_in,
                              void* d_out, int out_size, void* d_ws, size_t ws_size,
                              hipStream_t stream) {
  const int*   edge_index = (const int*)d_in[0];
  const int*   kinds      = (const int*)d_in[1];
  const float* edge_attr  = (const float*)d_in[2];
  const float* emb        = (const float*)d_in[3];
  const float* msg_w1     = (const float*)d_in[4];
  const float* msg_b1     = (const float*)d_in[5];
  const float* msg_w2     = (const float*)d_in[6];
  const float* msg_b2     = (const float*)d_in[7];
  const float* gru_wih    = (const float*)d_in[8];
  const float* gru_whh    = (const float*)d_in[9];
  const float* gru_bih    = (const float*)d_in[10];
  const float* gru_bhh    = (const float*)d_in[11];
  const float* head_w1    = (const float*)d_in[12];
  const float* head_b1    = (const float*)d_in[13];
  const float* head_w2    = (const float*)d_in[14];
  const float* head_b2    = (const float*)d_in[15];
  float* out = (float*)d_out;

  const int E = in_sizes[0] / 2;
  const int N = in_sizes[1];

  // Workspace carve-out (256B aligned)
  char* ws = (char*)d_ws;
  size_t off = 0;
  auto carve = [&](size_t bytes) {
    void* p = ws + off;
    off = (off + bytes + 255) & ~(size_t)255;
    return p;
  };
  float*  h_buf = (float*)carve((size_t)N * 64 * sizeof(float));
  float*  agg   = (float*)carve((size_t)N * 64 * sizeof(float));
  __bf16* w1f   = (__bf16*)carve((size_t)NLAYER * 4 * 5 * 32 * 16 * 2);
  __bf16* w2f   = (__bf16*)carve((size_t)NLAYER * 4 * 2 * 32 * 16 * 2);
  __bf16* hwf   = (__bf16*)carve((size_t)4 * 5 * 32 * 16 * 2);
  (void)ws_size; (void)n_in; (void)out_size;

  const int prep_total = NLAYER * 4 * 5 * 512 + NLAYER * 4 * 2 * 512 + 4 * 5 * 512;
  prep_weights_kernel<<<(prep_total + 255) / 256, 256, 0, stream>>>(
      msg_w1, msg_w2, head_w1, w1f, w2f, hwf);

  const int nh = N * 64;
  init_h_kernel<<<(nh + 255) / 256, 256, 0, stream>>>(h_buf, emb, kinds, nh);

  const int ntiles  = (E + 15) / 16;
  const int eblocks = (ntiles + 7) / 8;

  for (int l = 0; l < NLAYER; ++l) {
    zero_kernel<<<(nh + 255) / 256, 256, 0, stream>>>(agg, nh);
    edge_msg_kernel<<<eblocks, 256, 0, stream>>>(
        h_buf, edge_index, edge_attr, agg,
        w1f + (size_t)l * 4 * 5 * 512, msg_b1 + l * 64,
        w2f + (size_t)l * 4 * 2 * 512, msg_b2 + l * 64, E);
    gru_kernel<<<(N + 3) / 4, 256, 0, stream>>>(
        h_buf, agg,
        gru_wih + (size_t)l * 192 * 64, gru_whh + (size_t)l * 192 * 64,
        gru_bih + l * 192, gru_bhh + l * 192, N);
  }

  edge_head_kernel<<<eblocks, 256, 0, stream>>>(
      h_buf, edge_index, edge_attr, out, hwf, head_b1, head_w2, head_b2, E);
}
